// LLaMA_61340722921876
// MI455X (gfx1250) — matrix operand order repaired and checked by reading, not compiled, not run on hardware
//
#include <hip/hip_runtime.h>
#include <hip/hip_bf16.h>

typedef __bf16 bf16;
typedef __attribute__((ext_vector_type(16))) __bf16 v16bf;
typedef __attribute__((ext_vector_type(8)))  float  v8f;

constexpr int Dm   = 1024;   // model dim
constexpr int Hn   = 16;     // heads
constexpr int HDn  = 64;     // head dim
constexpr int DFFn = 4096;   // ffn hidden (pre-split)
constexpr int NLn  = 4;      // layers
constexpr int Vn   = 32000;  // vocab
constexpr int BSn  = 2048;   // B*S rows
constexpr int SEQn = 1024;   // sequence length

static __device__ __forceinline__ unsigned pack2(float a, float b) {
  union { bf16 h[2]; unsigned u; } p;
  p.h[0] = (bf16)a; p.h[1] = (bf16)b;
  return p.u;
}

// ---------------------------------------------------------------------------
// Embedding gather: x[row][d] = emb[tokens[row]][d]
// ---------------------------------------------------------------------------
__global__ void embed_kernel(const int* __restrict__ tokens,
                             const float* __restrict__ emb,
                             float* __restrict__ x) {
  int idx = blockIdx.x * 256 + threadIdx.x;          // over BSn*Dm
  int row = idx >> 10, d = idx & (Dm - 1);
  x[idx] = emb[(size_t)tokens[row] * Dm + d];
}

// ---------------------------------------------------------------------------
// RoPE, faithful to reference: pos m = s+1, theta_i = 10000^(-2i/D) (i=1..D/2),
// cos/sin tiled twice, rot = concat(-x_odd, x_even)
// ---------------------------------------------------------------------------
__global__ void rope_kernel(const float* __restrict__ x, float* __restrict__ o) {
  int idx = blockIdx.x * 256 + threadIdx.x;          // over BSn*Dm
  int row = idx >> 10, f = idx & (Dm - 1);
  int s = row & (SEQn - 1);
  float m = (float)(s + 1);
  float i = (float)((f & (Dm / 2 - 1)) + 1);
  float ang = m * __powf(10000.f, -2.f * i / (float)Dm);
  float c = __cosf(ang), sn = __sinf(ang);
  float rot = (f < Dm / 2) ? -x[(size_t)row * Dm + 2 * f + 1]
                           :  x[(size_t)row * Dm + 2 * (f - Dm / 2)];
  o[idx] = x[idx] * c + rot * sn;
}

// ---------------------------------------------------------------------------
// RMSNorm: one block per row
// ---------------------------------------------------------------------------
__global__ void rmsnorm_kernel(const float* __restrict__ x,
                               const float* __restrict__ scale,
                               float* __restrict__ o) {
  __shared__ float red[256];
  int row = blockIdx.x, tid = threadIdx.x;
  float s = 0.f;
  for (int c = tid; c < Dm; c += 256) {
    float v = x[(size_t)row * Dm + c];
    s += v * v;
  }
  red[tid] = s; __syncthreads();
  for (int off = 128; off > 0; off >>= 1) {
    if (tid < off) red[tid] += red[tid + off];
    __syncthreads();
  }
  float r = rsqrtf(red[0] / (float)Dm + 1e-5f);
  for (int c = tid; c < Dm; c += 256)
    o[(size_t)row * Dm + c] = scale[c] * x[(size_t)row * Dm + c] * r;
}

// ---------------------------------------------------------------------------
// SwiGLU + ReLU: g = max(a * silu(b), 0), a=h[:, :DFF/2], b=h[:, DFF/2:]
// ---------------------------------------------------------------------------
__global__ void swiglu_kernel(const float* __restrict__ h, float* __restrict__ g) {
  int idx = blockIdx.x * 256 + threadIdx.x;          // over BSn*(DFFn/2)
  int row = idx >> 11, c = idx & (DFFn / 2 - 1);
  float a = h[(size_t)row * DFFn + c];
  float b = h[(size_t)row * DFFn + DFFn / 2 + c];
  float v = a * (b / (1.f + __expf(-b)));
  g[idx] = v > 0.f ? v : 0.f;
}

// ---------------------------------------------------------------------------
// Tiled bf16-WMMA GEMM:  C[M,N] = A[M,K] @ B[K,N] + bias (+ res)
// Block tile 64x128, K-step 32, 8 waves; wave computes 32x32 (2x2 WMMA tiles).
// A tile staged global->LDS with GLOBAL_LOAD_ASYNC_TO_LDS_B128 (fp32, no VGPR
// round-trip; cvt to bf16 during fragment build). B tile needs a K-transpose,
// so it goes through VGPRs (cvt + transposed bf16 store) with a prefetch of
// the next K-step's rows.
// EPI: 0 = +bias, 1 = +bias+residual
// ---------------------------------------------------------------------------
template <int EPI>
__global__ __launch_bounds__(256) void gemm_bf16(
    const float* __restrict__ A, const float* __restrict__ B,
    const float* __restrict__ bias, const float* __restrict__ res,
    float* __restrict__ C, int M, int N, int K) {
  __shared__ __align__(16) float Asf[64][36];   // fp32 A tile (async-staged)
  __shared__ __align__(16) bf16 Bs[128][40];    // bf16, K-transposed: Bs[n][k]

  const int tid  = threadIdx.x;
  const int lane = tid & 31;
  const int wave = tid >> 5;
  const int half = lane >> 4;
  const int l16  = lane & 15;
  const int wm   = wave >> 2;     // 0..1 (M)
  const int wn   = wave & 3;      // 0..3 (N)
  const int m0   = blockIdx.y * 64;
  const int n0   = blockIdx.x * 128;

  const v8f vzero = {0.f, 0.f, 0.f, 0.f, 0.f, 0.f, 0.f, 0.f};
  v8f acc[2][2];
  acc[0][0] = vzero; acc[0][1] = vzero; acc[1][0] = vzero; acc[1][1] = vzero;

  for (int kb = 0; kb < K; kb += 32) {
    // --- A tile: 64x32 fp32 = 512 x 16B async transfers (2 per thread) ---
#pragma unroll
    for (int it = 0; it < 2; ++it) {
      int t = tid + it * 256;           // transfer id 0..511
      int r = t >> 3;                   // row 0..63
      int c = (t & 7) * 4;              // col 0,4,..,28
      unsigned lds_off = (unsigned)(uintptr_t)&Asf[r][c];
      const float* gp = A + (size_t)(m0 + r) * K + kb + c;
      asm volatile("global_load_async_to_lds_b128 %0, %1, off"
                   :: "v"(lds_off), "v"(gp) : "memory");
    }
    // --- B tile: cvt + transposed store; prefetch next K-step rows ---
    for (int i = tid; i < 32 * 128; i += 256) {
      int r = i >> 7, c = i & 127;
      const float* bp = B + (size_t)(kb + r) * N + n0 + c;
      Bs[c][r] = (bf16)(*bp);
      if (kb + 32 < K) __builtin_prefetch(bp + (size_t)32 * N, 0, 1);
    }
    asm volatile("s_wait_asynccnt 0" ::: "memory");
    __syncthreads();

    v16bf af[2], bfv[2];
#pragma unroll
    for (int t = 0; t < 2; ++t) {
      unsigned* au = reinterpret_cast<unsigned*>(&af[t]);
      int row = wm * 32 + t * 16 + l16;
#pragma unroll
      for (int p = 0; p < 8; ++p) {
        int k = half * 16 + 2 * p;
        const float* ap = &Asf[row][k];
        au[p] = pack2(ap[0], ap[1]);    // -> v_cvt_pk_bf16_f32
      }
      unsigned* bu = reinterpret_cast<unsigned*>(&bfv[t]);
      int col = wn * 32 + t * 16 + l16;
#pragma unroll
      for (int p = 0; p < 8; ++p)
        bu[p] = *reinterpret_cast<const unsigned*>(&Bs[col][half * 16 + 2 * p]);
    }
#pragma unroll
    for (int i = 0; i < 2; ++i)
#pragma unroll
      for (int j = 0; j < 2; ++j)
        acc[i][j] = __builtin_amdgcn_wmma_f32_16x16x32_bf16(
            false, af[i], false, bfv[j], (short)0, acc[i][j], false, false);
    __syncthreads();
  }

#pragma unroll
  for (int i = 0; i < 2; ++i)
#pragma unroll
    for (int j = 0; j < 2; ++j)
#pragma unroll
      for (int r = 0; r < 8; ++r) {
        int m = m0 + wm * 32 + i * 16 + half * 8 + r;
        int n = n0 + wn * 32 + j * 16 + l16;
        float v = acc[i][j][r] + bias[n];
        if (EPI == 1) v += res[(size_t)m * N + n];
        C[(size_t)m * N + n] = v;
      }
}

// ---------------------------------------------------------------------------
// Flash attention: grid (SEQ/64, H, B), 4 waves/block, 16 queries per wave,
// 32-key blocks (matches WMMA K), causal, online softmax.
// ---------------------------------------------------------------------------
__global__ __launch_bounds__(128) void attn_flash(
    const float* __restrict__ Q, const float* __restrict__ Kx,
    const float* __restrict__ Vx, float* __restrict__ O) {
  __shared__ __align__(16) bf16 Ps[4][16][40];   // per-wave P transpose buffer

  const int tid  = threadIdx.x;
  const int lane = tid & 31;
  const int wave = tid >> 5;
  const int half = lane >> 4;
  const int l16  = lane & 15;
  const int h    = blockIdx.y;
  const int b    = blockIdx.z;
  const int q0   = blockIdx.x * 64 + wave * 16;
  const size_t rowbase = (size_t)b * SEQn;       // token row offset
  const size_t hoff    = (size_t)h * HDn;

  // Q fragments for d=[0,32) and [32,64)
  v16bf qf[2];
#pragma unroll
  for (int t = 0; t < 2; ++t) {
    unsigned* au = reinterpret_cast<unsigned*>(&qf[t]);
    const float* qp = Q + (rowbase + q0 + l16) * Dm + hoff + t * 32;
#pragma unroll
    for (int p = 0; p < 8; ++p) {
      int k = half * 16 + 2 * p;
      au[p] = pack2(qp[k], qp[k + 1]);
    }
  }

  const v8f vzero = {0.f, 0.f, 0.f, 0.f, 0.f, 0.f, 0.f, 0.f};
  v8f oacc[4] = {vzero, vzero, vzero, vzero};
  float mrow[8], lrow[8];
#pragma unroll
  for (int r = 0; r < 8; ++r) { mrow[r] = -1e30f; lrow[r] = 0.f; }

  const int nkb = 2 * (blockIdx.x + 1);          // uniform across waves
  for (int kb = 0; kb < nkb; ++kb) {
    const int k0 = kb * 32;
    // S = Q @ K^T for this 16q x 32k block
    v8f s[2] = {vzero, vzero};
#pragma unroll
    for (int j = 0; j < 2; ++j) {
      const float* kp = Kx + (rowbase + k0 + j * 16 + l16) * Dm + hoff;
#pragma unroll
      for (int c = 0; c < 2; ++c) {
        v16bf kf;
        unsigned* ku = reinterpret_cast<unsigned*>(&kf);
#pragma unroll
        for (int p = 0; p < 8; ++p) {
          int d = c * 32 + half * 16 + 2 * p;
          ku[p] = pack2(kp[d], kp[d + 1]);
        }
        s[j] = __builtin_amdgcn_wmma_f32_16x16x32_bf16(
            false, qf[c], false, kf, (short)0, s[j], false, false);
      }
    }
    // scale, causal mask, online softmax
    float pv0[8], pv1[8];
#pragma unroll
    for (int r = 0; r < 8; ++r) {
      int qg = q0 + half * 8 + r;
      float v0 = s[0][r] * 0.125f;               // 1/sqrt(64)
      float v1 = s[1][r] * 0.125f;
      if (k0 + l16 > qg)      v0 = -1e30f;
      if (k0 + 16 + l16 > qg) v1 = -1e30f;
      float mx = fmaxf(v0, v1);
#pragma unroll
      for (int o = 1; o < 16; o <<= 1) mx = fmaxf(mx, __shfl_xor(mx, o, 32));
      float mnew  = fmaxf(mrow[r], mx);
      float alpha = __expf(mrow[r] - mnew);
      float p0 = __expf(v0 - mnew);
      float p1 = __expf(v1 - mnew);
      float rs = p0 + p1;
#pragma unroll
      for (int o = 1; o < 16; o <<= 1) rs += __shfl_xor(rs, o, 32);
      lrow[r] = lrow[r] * alpha + rs;
      mrow[r] = mnew;
      pv0[r] = p0; pv1[r] = p1;
#pragma unroll
      for (int t = 0; t < 4; ++t) oacc[t][r] *= alpha;
    }
    // transpose P from C/D layout to A layout via LDS
    __syncthreads();
#pragma unroll
    for (int r = 0; r < 8; ++r) {
      Ps[wave][half * 8 + r][l16]      = (bf16)pv0[r];
      Ps[wave][half * 8 + r][16 + l16] = (bf16)pv1[r];
    }
    __syncthreads();
    v16bf pf;
    unsigned* pu = reinterpret_cast<unsigned*>(&pf);
#pragma unroll
    for (int p = 0; p < 8; ++p) {
      int k = half * 16 + 2 * p;
      pu[p] = *reinterpret_cast<const unsigned*>(&Ps[wave][l16][k]);
    }
    // O += P @ V  (4 tiles across HD=64)
#pragma unroll
    for (int t = 0; t < 4; ++t) {
      v16bf vf;
      unsigned* vu = reinterpret_cast<unsigned*>(&vf);
#pragma unroll
      for (int p = 0; p < 8; ++p) {
        int kk = half * 16 + 2 * p;
        float x0 = Vx[(rowbase + k0 + kk)     * Dm + hoff + t * 16 + l16];
        float x1 = Vx[(rowbase + k0 + kk + 1) * Dm + hoff + t * 16 + l16];
        vu[p] = pack2(x0, x1);
      }
      oacc[t] = __builtin_amdgcn_wmma_f32_16x16x32_bf16(
          false, pf, false, vf, (short)0, oacc[t], false, false);
    }
  }
  // normalize and store (layout [B,S,D], head-interleaved, ready for Wo GEMM)
#pragma unroll
  for (int t = 0; t < 4; ++t)
#pragma unroll
    for (int r = 0; r < 8; ++r) {
      int qg = q0 + half * 8 + r;
      O[(rowbase + qg) * Dm + hoff + t * 16 + l16] = oacc[t][r] / lrow[r];
    }
}

// ---------------------------------------------------------------------------
// Host orchestration
// ---------------------------------------------------------------------------
extern "C" void kernel_launch(void* const* d_in, const int* in_sizes, int n_in,
                              void* d_out, int out_size, void* d_ws, size_t ws_size,
                              hipStream_t stream) {
  (void)in_sizes; (void)n_in; (void)out_size; (void)ws_size;
  const int*   tokens = (const int*)  d_in[0];
  const float* emb = (const float*)d_in[1];
  const float* Wq  = (const float*)d_in[2];
  const float* bq  = (const float*)d_in[3];
  const float* Wk  = (const float*)d_in[4];
  const float* bk  = (const float*)d_in[5];
  const float* Wv  = (const float*)d_in[6];
  const float* bv  = (const float*)d_in[7];
  const float* Wo  = (const float*)d_in[8];
  const float* bo  = (const float*)d_in[9];
  const float* W1  = (const float*)d_in[10];
  const float* b1  = (const float*)d_in[11];
  const float* W2  = (const float*)d_in[12];
  const float* b2  = (const float*)d_in[13];
  const float* s1  = (const float*)d_in[14];
  const float* s2  = (const float*)d_in[15];
  const float* sf  = (const float*)d_in[16];
  const float* Wh  = (const float*)d_in[17];
  const float* bh  = (const float*)d_in[18];
  float* out = (float*)d_out;

  float* x  = (float*)d_ws;                 // [BS, D]
  float* rx = x  + (size_t)BSn * Dm;        // [BS, D]
  float* n1 = rx + (size_t)BSn * Dm;        // [BS, D] (norm buffer, reused)
  float* qb = n1 + (size_t)BSn * Dm;        // [BS, D]
  float* kb = qb + (size_t)BSn * Dm;        // [BS, D]
  float* vb = kb + (size_t)BSn * Dm;        // [BS, D]
  float* ao = vb + (size_t)BSn * Dm;        // [BS, D]
  float* o1 = ao + (size_t)BSn * Dm;        // [BS, D]
  float* hf = o1 + (size_t)BSn * Dm;        // [BS, DFF]
  float* gf = hf + (size_t)BSn * DFFn;      // [BS, DFF/2]

  const dim3 gdd(Dm / 128, BSn / 64);

  embed_kernel<<<BSn * Dm / 256, 256, 0, stream>>>(tokens, emb, x);
  for (int l = 0; l < NLn; ++l) {
    rope_kernel<<<BSn * Dm / 256, 256, 0, stream>>>(x, rx);
    rmsnorm_kernel<<<BSn, 256, 0, stream>>>(rx, s1 + (size_t)l * Dm, n1);
    gemm_bf16<0><<<gdd, 256, 0, stream>>>(n1, Wq + (size_t)l * Dm * Dm,
                                          bq + (size_t)l * Dm, nullptr, qb, BSn, Dm, Dm);
    gemm_bf16<0><<<gdd, 256, 0, stream>>>(n1, Wk + (size_t)l * Dm * Dm,
                                          bk + (size_t)l * Dm, nullptr, kb, BSn, Dm, Dm);
    gemm_bf16<0><<<gdd, 256, 0, stream>>>(n1, Wv + (size_t)l * Dm * Dm,
                                          bv + (size_t)l * Dm, nullptr, vb, BSn, Dm, Dm);
    attn_flash<<<dim3(SEQn / 64, Hn, 2), 128, 0, stream>>>(qb, kb, vb, ao);
    gemm_bf16<1><<<gdd, 256, 0, stream>>>(ao, Wo + (size_t)l * Dm * Dm,
                                          bo + (size_t)l * Dm, rx, o1, BSn, Dm, Dm);
    rmsnorm_kernel<<<BSn, 256, 0, stream>>>(o1, s2 + (size_t)l * Dm, n1);
    gemm_bf16<0><<<dim3(DFFn / 128, BSn / 64), 256, 0, stream>>>(
        n1, W1 + (size_t)l * Dm * DFFn, b1 + (size_t)l * DFFn, nullptr, hf, BSn, DFFn, Dm);
    swiglu_kernel<<<BSn * (DFFn / 2) / 256, 256, 0, stream>>>(hf, gf);
    gemm_bf16<1><<<gdd, 256, 0, stream>>>(gf, W2 + (size_t)l * (DFFn / 2) * Dm,
                                          b2 + (size_t)l * Dm, o1, x, BSn, Dm, DFFn / 2);
  }
  rmsnorm_kernel<<<BSn, 256, 0, stream>>>(x, sf, n1);
  gemm_bf16<0><<<dim3(Vn / 128, BSn / 64), 256, 0, stream>>>(
      n1, Wh, bh, nullptr, out, BSn, Vn, Dm);
}
